// MultiHeadPosAtt_71382356460015
// MI455X (gfx1250) — matrix operand order repaired and checked by reading, not compiled, not run on hardware
//
#include <hip/hip_runtime.h>
#include <hip/hip_bf16.h>
#include <math.h>

#define N_HEADS 8
#define HIDDEN  256
#define VDIM    32
#define NQS     2048
#define NKS     2048
#define NBATCH  16
#define BVCOLS  (NBATCH * VDIM)   // 512 columns in the fused att@value GEMM

typedef __attribute__((ext_vector_type(16))) __bf16 v16bf;
typedef __attribute__((ext_vector_type(8)))  float  v8f;
typedef unsigned int v4u __attribute__((ext_vector_type(4)));
typedef int v8i __attribute__((ext_vector_type(8)));
typedef int v4i __attribute__((ext_vector_type(4)));

union V16U { v16bf v; uint4 q[2]; };   // 32B: two 16B global/LDS loads per WMMA operand
union Pack8 { uint4 q; unsigned short us[8]; };

__device__ __forceinline__ unsigned short f2bf(float f) {
    unsigned u = __float_as_uint(f);
    unsigned r = u + 0x7FFFu + ((u >> 16) & 1u);   // round-to-nearest-even
    return (unsigned short)(r >> 16);
}

// ---------------------------------------------------------------------------
// TDM: issue a 2D tile load (128 rows x 32 bf16, row stride NKS elements) from
// global into LDS. D# built per CDNA5 ISA:
//   group0: count=1 | lds_addr[63:32] | global_addr[120:64] | type=2
//   group1: data_size=2B, tensor_dim0=NKS, tensor_dim1=128,
//           tile_dim0=32, tile_dim1=128, tensor_dim0_stride=NKS
// Tracked with TENSORcnt; completion via s_wait_tensorcnt.
// ---------------------------------------------------------------------------
__device__ __forceinline__ void tdm_load_b_slab(unsigned lds_addr, unsigned long long gaddr) {
    v4u g0 = { 1u,                                   // count = 1 valid descriptor
               lds_addr,                              // LDS byte address (bits 63:32)
               (unsigned)gaddr,                       // global_addr[31:0]
               ((unsigned)(gaddr >> 32) & 0x1FFFFFFu) // global_addr[56:32]
                   | (2u << 30) };                    // type = 2 ("image")
    v8i g1 = { (int)(1u << 16),          // data_size = 1 -> 2 bytes/element
               (int)((unsigned)NKS << 16), // tensor_dim0[15:0] in word1[31:16]
               (int)(128u << 16),        // tensor_dim1[15:0] in word2[31:16]
               (int)(32u << 16),         // tile_dim0 in word3[31:16]
               128,                      // tile_dim1 in word4[15:0]
               NKS,                      // tensor_dim0_stride[31:0]
               0, 0 };
    v4i z4 = { 0, 0, 0, 0 };
#if __clang_major__ >= 23
    v8i z8 = { 0, 0, 0, 0, 0, 0, 0, 0 };
    __builtin_amdgcn_tensor_load_to_lds(g0, g1, z4, z4, z8, 0);
#else
    __builtin_amdgcn_tensor_load_to_lds(g0, g1, z4, z4, 0);
#endif
}

// ---------------------------------------------------------------------------
// Kernel 0a/0b: fp32 -> bf16 conversions (x row-major; weight transposed so the
// WMMA B-operand reads contiguous K per lane).
// ---------------------------------------------------------------------------
__global__ __launch_bounds__(256) void cvt_x_kernel(const float* __restrict__ x,
                                                    unsigned short* __restrict__ xbf,
                                                    int n) {
    int i = blockIdx.x * 256 + threadIdx.x;
    if (i < n) xbf[i] = f2bf(x[i]);
}

__global__ __launch_bounds__(256) void cvt_w_kernel(const float* __restrict__ w,
                                                    unsigned short* __restrict__ wT) {
    int i = blockIdx.x * 256 + threadIdx.x;      // i = ((h*HIDDEN)+j)*VDIM + v
    int v = i & (VDIM - 1);
    int j = (i >> 5) & (HIDDEN - 1);
    int h = i >> 13;
    wT[((size_t)(h * VDIM + v)) * HIDDEN + j] = f2bf(w[i]);
}

// ---------------------------------------------------------------------------
// Kernel 1: value = x @ weight  (per h,b: [NK x 256] @ [256 x 32]) via WMMA bf16.
// Output written transposed+packed: valueT[h][b*32+v][n]  (bf16) so the big GEMM
// reads its B operand as contiguous-K rows.
// ---------------------------------------------------------------------------
__global__ __launch_bounds__(256) void value_kernel(const unsigned short* __restrict__ xbf,
                                                    const unsigned short* __restrict__ wT,
                                                    unsigned short* __restrict__ valueT) {
    const int wave = threadIdx.x >> 5;
    const int lane = threadIdx.x & 31;
    const int blk  = blockIdx.x;            // (h*NBATCH + b)*16 + mchunk
    const int mchunk = blk & 15;
    const int hb = blk >> 4;
    const int b  = hb & (NBATCH - 1);
    const int h  = hb >> 4;
    const int mtile = mchunk * 8 + wave;
    const int nbase = mtile * 16;
    const int mrow  = lane & 15;
    const int c0 = (lane < 16) ? 0 : 8;     // unused; A loads use kh, like B
    const int kh = (lane < 16) ? 0 : 16;    // B-operand K half       (contiguous k = 16h + s, for A too)

    v8f acc0 = {};
    v8f acc1 = {};
    const unsigned short* arow  = xbf + ((size_t)(b * NKS + nbase + mrow)) * HIDDEN;
    const unsigned short* b0row = wT  + ((size_t)(h * VDIM + mrow)) * HIDDEN;
    const unsigned short* b1row = wT  + ((size_t)(h * VDIM + 16 + mrow)) * HIDDEN;

    for (int j = 0; j < HIDDEN; j += 32) {
        V16U a;
        a.q[0] = *(const uint4*)(arow + j + kh);
        a.q[1] = *(const uint4*)(arow + j + kh + 8);
        V16U b0, b1;
        b0.q[0] = *(const uint4*)(b0row + j + kh);
        b0.q[1] = *(const uint4*)(b0row + j + kh + 8);
        b1.q[0] = *(const uint4*)(b1row + j + kh);
        b1.q[1] = *(const uint4*)(b1row + j + kh + 8);
        acc0 = __builtin_amdgcn_wmma_f32_16x16x32_bf16(false, a.v, false, b0.v,
                                                       (short)0, acc0, false, false);
        acc1 = __builtin_amdgcn_wmma_f32_16x16x32_bf16(false, a.v, false, b1.v,
                                                       (short)0, acc1, false, false);
    }

    // D layout: VGPR i <-> M = i + (lane<16?0:8); 8 consecutive n per lane -> one 16B store.
    const int noff = nbase + ((lane < 16) ? 0 : 8);
    {
        Pack8 p;
        #pragma unroll
        for (int vi = 0; vi < 8; vi++) p.us[vi] = f2bf(acc0[vi]);
        size_t row = (size_t)h * BVCOLS + b * VDIM + mrow;           // nt = 0
        *(uint4*)(valueT + row * NKS + noff) = p.q;
    }
    {
        Pack8 p;
        #pragma unroll
        for (int vi = 0; vi < 8; vi++) p.us[vi] = f2bf(acc1[vi]);
        size_t row = (size_t)h * BVCOLS + b * VDIM + 16 + mrow;      // nt = 1
        *(uint4*)(valueT + row * NKS + noff) = p.q;
    }
}

// ---------------------------------------------------------------------------
// Kernel 2: per (h,q) row — scale, exact median threshold (order stats 1023/1024
// via bisection on float bit patterns; all values >= 0 so bit order == value
// order), masked softmax of -scaled, bf16 att row out.
// ---------------------------------------------------------------------------
__global__ __launch_bounds__(256) void att_kernel(const float* __restrict__ m_dist,
                                                  const float* __restrict__ r,
                                                  unsigned short* __restrict__ att) {
    __shared__ float sred[256];
    __shared__ int   scnt;
    __shared__ float fsum;

    const int row = blockIdx.x;            // h*NQS + q
    const int h   = row / NQS;
    const int t   = threadIdx.x;

    const float scale = tanf(0.25f * 3.14159265358979323846f *
                             (1.0f - 1e-7f) * (1.0f + sinf(r[h])));

    const float* src = m_dist + (size_t)row * NKS;
    float4 f0 = ((const float4*)src)[t * 2 + 0];
    float4 f1 = ((const float4*)src)[t * 2 + 1];
    float loc[8] = {f0.x, f0.y, f0.z, f0.w, f1.x, f1.y, f1.z, f1.w};
    unsigned ub[8];
    float mn = 3.4e38f;
    #pragma unroll
    for (int i = 0; i < 8; i++) {
        loc[i] *= scale;
        ub[i] = __float_as_uint(loc[i]);
        mn = fminf(mn, loc[i]);
    }

    // row min (for softmax stabilization: max(-x) = -min(x))
    sred[t] = mn;
    __syncthreads();
    for (int s = 128; s > 0; s >>= 1) {
        if (t < s) sred[t] = fminf(sred[t], sred[t + s]);
        __syncthreads();
    }
    const float rowmin = sred[0];
    __syncthreads();

    // order statistics s[1023], s[1024] (0-indexed) -> linear-interp percentile(50)
    unsigned kbits[2];
    for (int sel = 0; sel < 2; sel++) {
        const unsigned k = 1023u + (unsigned)sel;
        unsigned lo = 0u, hi = 0xFFFFFFFFu;
        while (lo < hi) {                     // uniform across block
            unsigned mid = lo + ((hi - lo) >> 1);
            if (t == 0) scnt = 0;
            __syncthreads();
            int c = 0;
            #pragma unroll
            for (int i = 0; i < 8; i++) c += (ub[i] <= mid) ? 1 : 0;
            atomicAdd(&scnt, c);
            __syncthreads();
            unsigned total = (unsigned)scnt;
            __syncthreads();
            if (total >= k + 1u) hi = mid; else lo = mid + 1u;
        }
        kbits[sel] = lo;
    }
    const float thr = 0.5f * (__uint_as_float(kbits[0]) + __uint_as_float(kbits[1]));

    // masked softmax of -scaled
    if (t == 0) fsum = 0.0f;
    __syncthreads();
    float p[8];
    float lsum = 0.0f;
    #pragma unroll
    for (int i = 0; i < 8; i++) {
        float pi = (loc[i] <= thr) ? __expf(rowmin - loc[i]) : 0.0f;
        p[i] = pi;
        lsum += pi;
    }
    atomicAdd(&fsum, lsum);
    __syncthreads();
    const float inv = 1.0f / fsum;

    Pack8 pk;
    #pragma unroll
    for (int i = 0; i < 8; i++) pk.us[i] = f2bf(p[i] * inv);
    ((uint4*)(att + (size_t)row * NKS))[t] = pk.q;
}

// ---------------------------------------------------------------------------
// Kernel 3: out = gelu(att @ valueT^T) per head. Workgroup = 8 waves computes a
// 128(M=q) x 128(N=b*32+v) tile. B slabs (128x32 bf16 = 8 KB) are DMA'd by the
// Tensor Data Mover into double-buffered LDS so the TDM of step i+1 overlaps
// the WMMA of step i. Wave 0 issues the TDM (EXEC ignored, wave-level op) and
// gates with s_wait_tensorcnt; barriers publish the slab to all 8 waves.
// A operand streamed from global (+prefetch). GELU fused in epilogue.
// ---------------------------------------------------------------------------
__global__ __launch_bounds__(256) void out_kernel(const unsigned short* __restrict__ att,
                                                  const unsigned short* __restrict__ valueT,
                                                  float* __restrict__ out) {
    __shared__ __align__(16) unsigned short Bs[2][128 * 32];   // 2 x 8 KB

    const int t    = threadIdx.x;
    const int wave = t >> 5;
    const int lane = t & 31;
    const int mgroup = blockIdx.x;        // 16
    const int ngroup = blockIdx.y;        // 4
    const int h      = blockIdx.z;        // 8
    const int mtile  = mgroup * 128 + wave * 16;
    const int mrow   = lane & 15;
    const int c0 = (lane < 16) ? 0 : 8;
    const int kh = (lane < 16) ? 0 : 16;

    v8f acc[8];
    const v8f zero = {};
    #pragma unroll
    for (int i = 0; i < 8; i++) acc[i] = zero;

    const unsigned short* arow = att + ((size_t)h * NQS + mtile + mrow) * NKS;

    // Low 32 bits of a generic shared pointer = LDS byte address (ISA §10.2 LDS aperture).
    const unsigned lds0 = (unsigned)(size_t)(const void*)&Bs[0][0];
    const unsigned lds1 = (unsigned)(size_t)(const void*)&Bs[1][0];
    const unsigned long long gbase =
        (unsigned long long)(size_t)(const void*)(valueT +
            ((size_t)h * BVCOLS + ngroup * 128) * NKS);

    // Prologue: DMA the first B slab into buffer 0.
    if (wave == 0) tdm_load_b_slab(lds0, gbase);

    for (int kb = 0; kb < NKS; kb += 32) {
        const int cur = (kb >> 5) & 1;
        __syncthreads();   // all waves done reading buffer cur^1 (previous step)
        if (wave == 0) {
            if (kb + 32 < NKS) {
                tdm_load_b_slab(cur ? lds0 : lds1,
                                gbase + (unsigned long long)(kb + 32) * 2u);
                __builtin_amdgcn_s_wait_tensorcnt(1);   // older DMA (buffer cur) done
            } else {
                __builtin_amdgcn_s_wait_tensorcnt(0);   // last slab resident
            }
        }
        __syncthreads();   // publish buffer cur to all waves

        V16U a;
        a.q[0] = *(const uint4*)(arow + kb + kh);
        a.q[1] = *(const uint4*)(arow + kb + kh + 8);
        if (kb + 32 < NKS) __builtin_prefetch(arow + kb + 32, 0, 3);

        #pragma unroll
        for (int nt = 0; nt < 8; nt++) {
            V16U bt;
            const unsigned short* bp = &Bs[cur][(nt * 16 + mrow) * 32 + kh];
            bt.q[0] = *(const uint4*)bp;
            bt.q[1] = *(const uint4*)(bp + 8);
            acc[nt] = __builtin_amdgcn_wmma_f32_16x16x32_bf16(false, a.v, false, bt.v,
                                                              (short)0, acc[nt], false, false);
        }
    }

    // epilogue: gelu (tanh approx, jax.nn.gelu default) + scatter to [B, NQ, HID]
    #pragma unroll
    for (int nt = 0; nt < 8; nt++) {
        const int col = ngroup * 128 + nt * 16 + mrow;   // 0..511 = b*32 + v
        const int b = col >> 5;
        const int v = col & 31;
        #pragma unroll
        for (int vi = 0; vi < 8; vi++) {
            const int q = mtile + vi + ((lane < 16) ? 0 : 8);
            float xv = acc[nt][vi];
            float g = 0.5f * xv * (1.0f + tanhf(0.7978845608028654f *
                                                (xv + 0.044715f * xv * xv * xv)));
            out[((size_t)b * NQS + q) * HIDDEN + h * VDIM + v] = g;
        }
    }
}

// ---------------------------------------------------------------------------
extern "C" void kernel_launch(void* const* d_in, const int* in_sizes, int n_in,
                              void* d_out, int out_size, void* d_ws, size_t ws_size,
                              hipStream_t stream) {
    const float* m_dist = (const float*)d_in[0];   // [H, NQ, NK]
    const float* x      = (const float*)d_in[1];   // [B, NK, HID]
    const float* r      = (const float*)d_in[2];   // [H, 1, 1]
    const float* weight = (const float*)d_in[3];   // [H, HID, V]
    float* out          = (float*)d_out;           // [B, NQ, HID]

    char* ws = (char*)d_ws;
    const size_t xbf_bytes    = (size_t)NBATCH * NKS * HIDDEN * 2;   // 16 MB
    const size_t wT_bytes     = (size_t)N_HEADS * VDIM * HIDDEN * 2; // 128 KB
    const size_t valueT_bytes = (size_t)N_HEADS * BVCOLS * NKS * 2;  // 16 MB
    unsigned short* xbf    = (unsigned short*)(ws);
    unsigned short* wT     = (unsigned short*)(ws + xbf_bytes);
    unsigned short* valueT = (unsigned short*)(ws + xbf_bytes + wT_bytes);
    unsigned short* att    = (unsigned short*)(ws + xbf_bytes + wT_bytes + valueT_bytes); // 64 MB

    const int nx = NBATCH * NKS * HIDDEN;                 // 8.4M
    cvt_x_kernel<<<(nx + 255) / 256, 256, 0, stream>>>(x, xbf, nx);
    cvt_w_kernel<<<(N_HEADS * HIDDEN * VDIM) / 256, 256, 0, stream>>>(weight, wT);

    value_kernel<<<N_HEADS * NBATCH * 16, 256, 0, stream>>>(xbf, wT, valueT);
    att_kernel<<<N_HEADS * NQS, 256, 0, stream>>>(m_dist, r, att);
    out_kernel<<<dim3(NQS / 128, BVCOLS / 128, N_HEADS), 256, 0, stream>>>(att, valueT, out);
}
